// Decoder_40793599378067
// MI455X (gfx1250) — compile-verified, matrix operand order checked by reading, not run on hardware
//
#include <hip/hip_runtime.h>

// ---------------------------------------------------------------------------
// Problem constants
// ---------------------------------------------------------------------------
#define NF   128
#define LAT  256
#define IMGD 64
#define BB   8
#define NCAL 16
#define FIN  384           // NF*3
#define FH   128
#define FO   256
#define HW   4096          // 64*64
#define KTOT 197376

// segment offsets in kall
#define OFF_WIN   0
#define OFF_BIN   49152
#define OFF_WMID  49280
#define OFF_BMID  65664
#define OFF_WOUT  65792
#define OFF_BOUT  98560
#define OFF_WSH   98816
#define OFF_BSH   197120

typedef __attribute__((ext_vector_type(16))) __bf16 v16bf;
typedef __attribute__((ext_vector_type(8)))  __bf16 v8bf;
typedef __attribute__((ext_vector_type(8)))  float  v8f;
typedef __attribute__((ext_vector_type(4)))  unsigned int v4u;

// round-to-nearest-even f32 -> bf16 bit pattern
__device__ __forceinline__ unsigned short f2bf(float f) {
  unsigned int u = __float_as_uint(f);
  unsigned int r = (u + 0x7FFFu + ((u >> 16) & 1u)) >> 16;
  return (unsigned short)r;
}

// Fragment-blocked bf16 activation layout:
//   elem offset = ((((b*KB + kb)*256 + nb)*32 + lane)*16 + j)
// where for channel c: kb=c>>5, kr=c&31, half=(kr>>3)&1, j=((kr>>4)<<3)+(kr&7),
// lane = half*16 + (pix&15), nb = pix>>4.   (KB = C/32, 256 = HW/16)
__device__ __forceinline__ size_t frag_off(int b, int c, int pix, int KB) {
  int kb   = c >> 5;
  int kr   = c & 31;
  int half = (kr >> 3) & 1;
  int j    = ((kr >> 4) << 3) + (kr & 7);
  int nb   = pix >> 4;
  int lane = half * 16 + (pix & 15);
  return ((((size_t)b * KB + kb) * 256 + nb) * 32 + lane) * 16 + j;
}

// ---------------------------------------------------------------------------
// Hypernetwork GEMM: kall = lat @ hyper_w + hyper_b, fused scale + bf16 pack
// ---------------------------------------------------------------------------
__global__ __launch_bounds__(256) void hyper_kernel(
    const float* __restrict__ lat, const float* __restrict__ hw,
    const float* __restrict__ hb,
    unsigned short* __restrict__ w_in,  unsigned short* __restrict__ w_mid,
    unsigned short* __restrict__ w_out, unsigned short* __restrict__ w_sh,
    float* __restrict__ b_in,  float* __restrict__ b_mid,
    float* __restrict__ b_out, float* __restrict__ b_sh) {
  __shared__ float lats[BB * LAT];
  int t = threadIdx.x;
  for (int i = t; i < BB * LAT; i += 256) lats[i] = lat[i];
  __syncthreads();
  int k = blockIdx.x * 256 + t;
  if (k >= KTOT) return;
  float acc[BB];
  float bias = hb[k];
#pragma unroll
  for (int b = 0; b < BB; ++b) acc[b] = bias;
  for (int l = 0; l < LAT; ++l) {
    float w = hw[(size_t)l * KTOT + k];
#pragma unroll
    for (int b = 0; b < BB; ++b) acc[b] = fmaf(lats[b * LAT + l], w, acc[b]);
  }
  const float s384 = 0.051031036307982884f;  // 1/sqrt(384)
  const float s128 = 0.08838834764831845f;   // 1/sqrt(128)
#pragma unroll
  for (int b = 0; b < BB; ++b) {
    float v = acc[b];
    if (k < OFF_BIN)        w_in [(size_t)b*49152 + (k - OFF_WIN)]  = f2bf(v * s384);
    else if (k < OFF_WMID)  b_in [(size_t)b*128   + (k - OFF_BIN)]  = v;
    else if (k < OFF_BMID)  w_mid[(size_t)b*16384 + (k - OFF_WMID)] = f2bf(v * s128);
    else if (k < OFF_WOUT)  b_mid[(size_t)b*128   + (k - OFF_BMID)] = v;
    else if (k < OFF_BOUT)  w_out[(size_t)b*32768 + (k - OFF_WOUT)] = f2bf(v * s128);
    else if (k < OFF_WSH)   b_out[(size_t)b*256   + (k - OFF_BOUT)] = v;
    else if (k < OFF_BSH)   w_sh [(size_t)b*98304 + (k - OFF_WSH)]  = f2bf(v * s384);
    else                    b_sh [(size_t)b*256   + (k - OFF_BSH)]  = v;
  }
}

// ---------------------------------------------------------------------------
// init: state = ca_init; embs[0] = ca_init
// ---------------------------------------------------------------------------
__global__ void init_kernel(const float* __restrict__ ca,
                            float* __restrict__ state,
                            float* __restrict__ embs0) {
  for (size_t i = (size_t)blockIdx.x * blockDim.x + threadIdx.x;
       i < (size_t)BB * NF * HW; i += (size_t)gridDim.x * blockDim.x) {
    float v = ca[i];
    state[i] = v;
    embs0[i] = v;
  }
}

// ---------------------------------------------------------------------------
// prep: p = [out, sobel_x(out), sobel_y(out)], instance-norm, write bf16 frag
// one workgroup per (channel c, batch b)
// ---------------------------------------------------------------------------
__global__ __launch_bounds__(256) void prep_kernel(
    const float* __restrict__ state, unsigned short* __restrict__ Pfrag) {
  __shared__ float tile[68 * 68];
  __shared__ float red[6 * 256];
  int c = blockIdx.x, b = blockIdx.y, t = threadIdx.x;

  for (int i = t; i < 68 * 68; i += 256) tile[i] = 0.f;
  __syncthreads();
  const float* src = state + ((size_t)b * NF + c) * HW;
  for (int i = t; i < HW; i += 256) {
    int y = i >> 6, x = i & 63;
    tile[(y + 2) * 68 + (x + 2)] = src[i];
  }
  __syncthreads();

  // r = linspace(-pi/2, pi/2, 5): d=sin(r)={-1,-q,0,q,1}, s=cos(r)={0,q,1,q,0}
  const float q = 0.70710678118654752f;
  const float d0 = -1.f, d1 = -q, d3 = q, d4 = 1.f;
  const float s1 = q, s2 = 1.f, s3 = q;

  float v0[16], vx[16], vy[16];
  float sm[6] = {0.f, 0.f, 0.f, 0.f, 0.f, 0.f};
#pragma unroll
  for (int qq = 0; qq < 16; ++qq) {
    int i = t + qq * 256;
    int y = i >> 6, x = i & 63;
    const float* tp = &tile[y * 68 + x];
    float a = tp[2 * 68 + 2];
    // kx[i][j] = s_i * d_j  (rows 0,4 zero; col 2 zero)
    float rx = s1 * (d0*tp[1*68+0] + d1*tp[1*68+1] + d3*tp[1*68+3] + d4*tp[1*68+4])
             + s2 * (d0*tp[2*68+0] + d1*tp[2*68+1] + d3*tp[2*68+3] + d4*tp[2*68+4])
             + s3 * (d0*tp[3*68+0] + d1*tp[3*68+1] + d3*tp[3*68+3] + d4*tp[3*68+4]);
    // ky[i][j] = d_i * s_j  (cols 0,4 zero; row 2 zero)
    float ry = s1 * (d0*tp[0*68+1] + d1*tp[1*68+1] + d3*tp[3*68+1] + d4*tp[4*68+1])
             + s2 * (d0*tp[0*68+2] + d1*tp[1*68+2] + d3*tp[3*68+2] + d4*tp[4*68+2])
             + s3 * (d0*tp[0*68+3] + d1*tp[1*68+3] + d3*tp[3*68+3] + d4*tp[4*68+3]);
    v0[qq] = a;  vx[qq] = rx;  vy[qq] = ry;
    sm[0] += a;  sm[1] += a * a;
    sm[2] += rx; sm[3] += rx * rx;
    sm[4] += ry; sm[5] += ry * ry;
  }
#pragma unroll
  for (int j = 0; j < 6; ++j) red[j * 256 + t] = sm[j];
  __syncthreads();
  for (int s = 128; s > 0; s >>= 1) {
    if (t < s) {
#pragma unroll
      for (int j = 0; j < 6; ++j) red[j * 256 + t] += red[j * 256 + t + s];
    }
    __syncthreads();
  }
  const float inv = 1.f / 4096.f;
  float mu0 = red[0] * inv, mu1 = red[2 * 256] * inv, mu2 = red[4 * 256] * inv;
  float rs0 = rsqrtf(red[1 * 256] * inv - mu0 * mu0 + 1e-5f);
  float rs1 = rsqrtf(red[3 * 256] * inv - mu1 * mu1 + 1e-5f);
  float rs2 = rsqrtf(red[5 * 256] * inv - mu2 * mu2 + 1e-5f);

  const int KB = FIN / 32;  // 12
#pragma unroll
  for (int qq = 0; qq < 16; ++qq) {
    int pix = t + qq * 256;
    Pfrag[frag_off(b, c,        pix, KB)] = f2bf((v0[qq] - mu0) * rs0);
    Pfrag[frag_off(b, c + 128,  pix, KB)] = f2bf((vx[qq] - mu1) * rs1);
    Pfrag[frag_off(b, c + 256,  pix, KB)] = f2bf((vy[qq] - mu2) * rs2);
  }
}

// ---------------------------------------------------------------------------
// WMMA bf16 GEMM core:  D[o,pix] = sum_c W[o,c] * Bfrag[c,pix]
// Each wave owns a 32(M) x 64(N) tile: 2 A-frags x 4 B-frags -> 8 WMMAs
// per K-step from only 6 fragment loads (vs 9 for a 16x128 strip).
// ---------------------------------------------------------------------------
__device__ __forceinline__ void gemm_acc(
    const __bf16* __restrict__ W, const __bf16* __restrict__ Bfrag,
    int K, int Mtot, int b, int Mbase2, int nbBase, int lane, v8f acc[2][4]) {
  const int KB = K >> 5;
  const int khalf = (lane >> 4) * 8;
  const int mlane = lane & 15;
  for (int kb = 0; kb < KB; ++kb) {
    const int kk = kb << 5;
    v16bf a[2];
#pragma unroll
    for (int mf = 0; mf < 2; ++mf) {
      const __bf16* wp = W + (size_t)b * Mtot * K +
                         (size_t)(Mbase2 + mf * 16 + mlane) * K + kk + khalf;
      v8bf lo = *(const v8bf*)wp;
      v8bf hi = *(const v8bf*)(wp + 16);
      a[mf] = __builtin_shufflevector(lo, hi, 0, 1, 2, 3, 4, 5, 6, 7,
                                              8, 9, 10, 11, 12, 13, 14, 15);
    }
    const __bf16* bp =
        Bfrag + ((((size_t)b * KB + kb) * 256 + (size_t)nbBase) * 32 + lane) * 16;
#pragma unroll
    for (int nf = 0; nf < 4; ++nf) {
      v16bf bv = *(const v16bf*)(bp + (size_t)nf * 512);
#pragma unroll
      for (int mf = 0; mf < 2; ++mf)
        acc[mf][nf] = __builtin_amdgcn_wmma_f32_16x16x32_bf16(
            false, a[mf], false, bv, (short)0, acc[mf][nf], false, false);
    }
  }
}

__global__ __launch_bounds__(256) void gemm_wmma(
    const unsigned short* __restrict__ W1, const unsigned short* __restrict__ B1, int K1,
    const unsigned short* __restrict__ W2, const unsigned short* __restrict__ B2, int K2,
    const float* __restrict__ bias1, const float* __restrict__ bias2,
    int Mtot, unsigned short* __restrict__ outFrag, float* __restrict__ outF32,
    int relu) {
  int lane = threadIdx.x & 31;
  int wave = threadIdx.x >> 5;
  int mgrp = wave & 3;          // 4 M-groups of 32 rows
  int ngrp = wave >> 2;         // 2 N-groups of 64 pixels
  int ntile = blockIdx.x;       // 0..31 -> 128 pixels each
  int b = blockIdx.y;
  int Mbase2 = blockIdx.z * 128 + mgrp * 32;
  int nbBase = ntile * 8 + ngrp * 4;

  const v8f vzero = {0.f, 0.f, 0.f, 0.f, 0.f, 0.f, 0.f, 0.f};
  v8f acc[2][4];
#pragma unroll
  for (int mf = 0; mf < 2; ++mf)
#pragma unroll
    for (int nf = 0; nf < 4; ++nf) acc[mf][nf] = vzero;

  gemm_acc((const __bf16*)W1, (const __bf16*)B1, K1, Mtot, b, Mbase2, nbBase,
           lane, acc);
  if (W2)
    gemm_acc((const __bf16*)W2, (const __bf16*)B2, K2, Mtot, b, Mbase2, nbBase,
             lane, acc);

#pragma unroll
  for (int mf = 0; mf < 2; ++mf) {
    int rbase = Mbase2 + mf * 16 + (lane >> 4) * 8;  // 8 consecutive M rows
#pragma unroll
    for (int r = 0; r < 8; ++r) {
      float bv = bias1[(size_t)b * Mtot + rbase + r];
      if (bias2) bv += bias2[(size_t)b * Mtot + rbase + r];
#pragma unroll
      for (int nf = 0; nf < 4; ++nf) {
        float v = acc[mf][nf][r] + bv;
        if (relu) v = fmaxf(v, 0.f);
        acc[mf][nf][r] = v;
      }
    }

    if (outFrag) {  // write bf16 in fragment layout for next GEMM (K = Mtot)
      int KB2  = Mtot >> 5;
      int kb2  = rbase >> 5;
      int kr   = rbase & 31;               // in {0,8,16,24}
      int half = (kr >> 3) & 1;
      int jb   = (kr >> 4) << 3;           // 0 or 8
      int dlane = half * 16 + (lane & 15);
#pragma unroll
      for (int nf = 0; nf < 4; ++nf) {
        int nb = nbBase + nf;
        v4u pk;
#pragma unroll
        for (int i = 0; i < 4; ++i)
          pk[i] = (unsigned int)f2bf(acc[mf][nf][2 * i]) |
                  ((unsigned int)f2bf(acc[mf][nf][2 * i + 1]) << 16);
        unsigned short* dp =
            outFrag + ((((size_t)b * KB2 + kb2) * 256 + nb) * 32 + dlane) * 16 + jb;
        *(v4u*)dp = pk;
      }
    }
    if (outF32) {  // plain [b][o][pix] fp32 (dn buffer)
#pragma unroll
      for (int nf = 0; nf < 4; ++nf) {
        int pix = (nbBase + nf) * 16 + (lane & 15);
        float* dp = outF32 + ((size_t)b * Mtot + rbase) * HW + pix;
#pragma unroll
        for (int r = 0; r < 8; ++r) dp[(size_t)r * HW] = acc[mf][nf][r];
      }
    }
  }
}

// ---------------------------------------------------------------------------
// update: out += leak * (val * sigmoid(gate)); store state + emb slot
// ---------------------------------------------------------------------------
__global__ void update_kernel(const float* __restrict__ dn,
                              float* __restrict__ state,
                              float* __restrict__ emb,
                              const float* __restrict__ leakp) {
  float lk = fminf(fmaxf(leakp[0], 0.001f), 1000.f);
  for (size_t i = (size_t)blockIdx.x * blockDim.x + threadIdx.x;
       i < (size_t)BB * NF * HW; i += (size_t)gridDim.x * blockDim.x) {
    int pix = (int)(i & 4095);
    int r = (int)(i >> 12);
    int c = r & 127;
    int b = r >> 7;
    float val  = dn[(((size_t)b * FO + c) << 12) + pix];
    float gate = dn[(((size_t)b * FO + 128 + c) << 12) + pix];
    float nv = val / (1.f + __expf(-gate));
    float o = state[i] + lk * nv;
    state[i] = o;
    emb[i] = o;
  }
}

// ---------------------------------------------------------------------------
// generic fp32 3x3 conv, pad 1, optional relu / residual / clamp copy
// ---------------------------------------------------------------------------
__global__ void conv3x3_k(const float* __restrict__ x, const float* __restrict__ w,
                          const float* __restrict__ bias, const float* __restrict__ res,
                          float* __restrict__ out, float* __restrict__ outClamp,
                          int Cin, int Cout, int relu) {
  size_t total = (size_t)BB * Cout * HW;
  for (size_t idx = (size_t)blockIdx.x * blockDim.x + threadIdx.x;
       idx < total; idx += (size_t)gridDim.x * blockDim.x) {
    int pix = (int)(idx & 4095);
    int px = pix & 63, py = pix >> 6;
    int rest = (int)(idx >> 12);
    int o = rest % Cout;
    int b = rest / Cout;
    float acc = bias[o];
    for (int c = 0; c < Cin; ++c) {
      const float* xp = x + ((size_t)b * Cin + c) * HW;
      const float* wp = w + ((size_t)o * Cin + c) * 9;
#pragma unroll
      for (int dy = 0; dy < 3; ++dy) {
        int yy = py + dy - 1;
        if (yy < 0 || yy > 63) continue;
#pragma unroll
        for (int dx = 0; dx < 3; ++dx) {
          int xx = px + dx - 1;
          if (xx < 0 || xx > 63) continue;
          acc = fmaf(wp[dy * 3 + dx], xp[yy * 64 + xx], acc);
        }
      }
    }
    if (relu) acc = fmaxf(acc, 0.f);
    if (res) acc += res[idx];
    out[idx] = acc;
    if (outClamp) outClamp[idx] = fminf(fmaxf(acc, -1.f), 1.f);
  }
}

// ---------------------------------------------------------------------------
// launch
// ---------------------------------------------------------------------------
extern "C" void kernel_launch(void* const* d_in, const int* in_sizes, int n_in,
                              void* d_out, int out_size, void* d_ws, size_t ws_size,
                              hipStream_t stream) {
  const float* lat  = (const float*)d_in[0];
  const float* ca   = (const float*)d_in[1];
  const float* leak = (const float*)d_in[2];
  const float* hw   = (const float*)d_in[3];
  const float* hb   = (const float*)d_in[4];
  const float* rw1  = (const float*)d_in[5];
  const float* rb1  = (const float*)d_in[6];
  const float* rw2  = (const float*)d_in[7];
  const float* rb2  = (const float*)d_in[8];
  const float* iw   = (const float*)d_in[9];
  const float* ib   = (const float*)d_in[10];

  float* out     = (float*)d_out;
  float* out_img = out;                                        // (8,3,64,64)
  float* embs    = out + (size_t)BB * 3 * HW;                  // (17,8,128,64,64)
  float* out_raw = embs + (size_t)(NCAL + 1) * BB * NF * HW;   // (8,3,64,64)

  char* ws = (char*)d_ws;
  size_t cur = 0;
  auto alloc = [&](size_t bytes) {
    void* p = ws + cur;
    cur = (cur + bytes + 255) & ~(size_t)255;
    return p;
  };

  float*          state = (float*)alloc((size_t)BB * NF * HW * 4);
  unsigned short* w_in  = (unsigned short*)alloc((size_t)BB * FH * FIN * 2);
  unsigned short* w_mid = (unsigned short*)alloc((size_t)BB * FH * FH * 2);
  unsigned short* w_out = (unsigned short*)alloc((size_t)BB * FO * FH * 2);
  unsigned short* w_sh  = (unsigned short*)alloc((size_t)BB * FO * FIN * 2);
  float*          b_in  = (float*)alloc((size_t)BB * FH * 4);
  float*          b_mid = (float*)alloc((size_t)BB * FH * 4);
  float*          b_out = (float*)alloc((size_t)BB * FO * 4);
  float*          b_sh  = (float*)alloc((size_t)BB * FO * 4);
  unsigned short* Pfrag = (unsigned short*)alloc((size_t)BB * FIN * HW * 2);
  unsigned short* H1    = (unsigned short*)alloc((size_t)BB * FH * HW * 2);
  unsigned short* H2    = (unsigned short*)alloc((size_t)BB * FH * HW * 2);
  float*          dn    = (float*)alloc((size_t)BB * FO * HW * 4);
  float* hbuf = (float*)Pfrag;   // reuse after CA loop (16.8MB <= 25.2MB)
  float* ybuf = (float*)dn;      // reuse after CA loop (16.8MB <= 33.6MB)

  hyper_kernel<<<(KTOT + 255) / 256, 256, 0, stream>>>(
      lat, hw, hb, w_in, w_mid, w_out, w_sh, b_in, b_mid, b_out, b_sh);
  init_kernel<<<4096, 256, 0, stream>>>(ca, state, embs);

  for (int it = 0; it < NCAL; ++it) {
    prep_kernel<<<dim3(NF, BB), 256, 0, stream>>>(state, Pfrag);
    // h1 = relu(W_in p + b_in)
    gemm_wmma<<<dim3(32, BB, 1), 256, 0, stream>>>(
        w_in, Pfrag, FIN, nullptr, nullptr, 0, b_in, nullptr, FH, H1, nullptr, 1);
    // h2 = relu(W_mid h1 + b_mid)
    gemm_wmma<<<dim3(32, BB, 1), 256, 0, stream>>>(
        w_mid, H1, FH, nullptr, nullptr, 0, b_mid, nullptr, FH, H2, nullptr, 1);
    // dn = W_out h2 + b_out + W_sh p + b_sh
    gemm_wmma<<<dim3(32, BB, 2), 256, 0, stream>>>(
        w_out, H2, FH, w_sh, Pfrag, FIN, b_out, b_sh, FO, nullptr, dn, 0);
    update_kernel<<<4096, 256, 0, stream>>>(
        dn, state, embs + (size_t)(it + 1) * BB * NF * HW, leak);
  }

  // h = relu(conv3x3(out, res_w1, res_b1))
  conv3x3_k<<<16384, 256, 0, stream>>>(state, rw1, rb1, nullptr, hbuf, nullptr,
                                       NF, NF, 1);
  // y = out + conv3x3(h, res_w2, res_b2)
  conv3x3_k<<<16384, 256, 0, stream>>>(hbuf, rw2, rb2, state, ybuf, nullptr,
                                       NF, NF, 0);
  // out_raw = conv3x3(y, img_w, img_b); out_img = clip(out_raw)
  conv3x3_k<<<384, 256, 0, stream>>>(ybuf, iw, ib, nullptr, out_raw, out_img,
                                     NF, 3, 0);
}
